// ANodeRnnCell_309237645638
// MI455X (gfx1250) — compile-verified, matrix operand order checked by reading, not run on hardware
//
#include <hip/hip_runtime.h>
#include <hip/hip_bf16.h>

// ---------------------------------------------------------------------------
// AGCRN cell for MI455X (gfx1250, wave32, WMMA bf16 16x16x32, f32 accumulate)
// B=64, N=4096, D_IN=D_OUT=64, E=16
// ---------------------------------------------------------------------------

typedef __hip_bfloat16 bf16_t;
typedef __attribute__((ext_vector_type(16))) __bf16 bfrag;   // 32B = 8 VGPRs
typedef __attribute__((ext_vector_type(8)))  float  v8f;     // f32 accumulator

union FragU { bfrag v; int4 q[2]; };

// Per-lane fragment loader for 16x32 bf16 A (row-major [m][k]) or B stored as
// Bt row-major [n][k].  ISA layout: lane = half*16 + row; VGPR0-3 hold
// k = half*8 + {0..7}, VGPR4-7 hold k = 16 + half*8 + {0..7}  -> two 16B loads.
__device__ __forceinline__ bfrag frag_load(const bf16_t* __restrict__ base, int ld) {
  const int lane = threadIdx.x & 31;
  const int row  = lane & 15;
  const int half = lane >> 4;
  const bf16_t* p = base + (size_t)row * ld + half * 8;
  FragU u;
  u.q[0] = *reinterpret_cast<const int4*>(p);
  u.q[1] = *reinterpret_cast<const int4*>(p + 16);
  return u.v;
}

__device__ __forceinline__ void frag_prefetch(const bf16_t* __restrict__ base, int ld) {
  const int lane = threadIdx.x & 31;
  __builtin_prefetch((const void*)(base + (size_t)(lane & 15) * ld), 0, 3);
}

__device__ __forceinline__ v8f wmma_bf16(bfrag a, bfrag b, v8f c) {
  // v_wmma_f32_16x16x32_bf16 : D = A x B + C, f32 accumulate
  return __builtin_amdgcn_wmma_f32_16x16x32_bf16(false, a, false, b, (short)0, c,
                                                 false, false);
}

// ---------------------------------------------------------------------------
// 1) supports = softmax(relu(E @ E^T)) row-wise, output bf16 (N x N, row-major)
// ---------------------------------------------------------------------------
__global__ void __launch_bounds__(256) k_supports(const float* __restrict__ emb,
                                                  bf16_t* __restrict__ S) {
  __shared__ float vals[4096];
  __shared__ float red[256];
  __shared__ float erow[16];
  const int n = blockIdx.x;
  const int tid = threadIdx.x;
  if (tid < 16) erow[tid] = emb[n * 16 + tid];
  __syncthreads();
  float e[16];
#pragma unroll
  for (int d = 0; d < 16; ++d) e[d] = erow[d];

  float lmax = 0.f;                              // relu output >= 0
  for (int m = tid; m < 4096; m += 256) {
    const float4* p = reinterpret_cast<const float4*>(emb + (size_t)m * 16);
    float4 q0 = p[0], q1 = p[1], q2 = p[2], q3 = p[3];
    float dot = e[0]*q0.x + e[1]*q0.y + e[2]*q0.z + e[3]*q0.w
              + e[4]*q1.x + e[5]*q1.y + e[6]*q1.z + e[7]*q1.w
              + e[8]*q2.x + e[9]*q2.y + e[10]*q2.z + e[11]*q2.w
              + e[12]*q3.x + e[13]*q3.y + e[14]*q3.z + e[15]*q3.w;
    float v = fmaxf(dot, 0.f);
    vals[m] = v;
    lmax = fmaxf(lmax, v);
  }
  red[tid] = lmax;
  __syncthreads();
  for (int s = 128; s > 0; s >>= 1) {
    if (tid < s) red[tid] = fmaxf(red[tid], red[tid + s]);
    __syncthreads();
  }
  const float gmax = red[0];
  __syncthreads();

  float lsum = 0.f;
  for (int m = tid; m < 4096; m += 256) {
    float ex = __expf(vals[m] - gmax);
    vals[m] = ex;
    lsum += ex;
  }
  red[tid] = lsum;
  __syncthreads();
  for (int s = 128; s > 0; s >>= 1) {
    if (tid < s) red[tid] += red[tid + s];
    __syncthreads();
  }
  const float inv = 1.f / red[0];
  for (int m = tid; m < 4096; m += 256)
    S[(size_t)n * 4096 + m] = __float2bfloat16(vals[m] * inv);
}

// ---------------------------------------------------------------------------
// 2) pack/transpose: dst[(b*64+c)][m] = bf16(src[b][m][c])   (GEMM B-operand)
// ---------------------------------------------------------------------------
__global__ void __launch_bounds__(256) k_pack(const float* __restrict__ src,
                                              bf16_t* __restrict__ dst) {
  __shared__ float tile[64][65];
  const int m0 = blockIdx.x * 64;
  const int b  = blockIdx.y;
  for (int idx = threadIdx.x; idx < 4096; idx += 256) {
    int r = idx >> 6, c = idx & 63;
    tile[r][c] = src[((size_t)b * 4096 + m0 + r) * 64 + c];
  }
  __syncthreads();
  for (int idx = threadIdx.x; idx < 4096; idx += 256) {
    int c = idx >> 6, r = idx & 63;
    dst[(size_t)(b * 64 + c) * 4096 + m0 + r] = __float2bfloat16(tile[r][c]);
  }
}

// ---------------------------------------------------------------------------
// 3) per-node weights: Wout[n][o][i] = bf16( sum_d E[n][d] * pool[d][i][o] )
//    Thread owns one (o,i); pool read from L2 exactly once; E via scalar loads.
// ---------------------------------------------------------------------------
__global__ void __launch_bounds__(256) k_wexpand(const float* __restrict__ emb,
                                                 const float* __restrict__ pool,
                                                 bf16_t* __restrict__ Wout,
                                                 int I, int O) {
  const int oi = blockIdx.x * 256 + threadIdx.x;
  const int o = oi / I, i = oi - o * I;
  float w[16];
#pragma unroll
  for (int d = 0; d < 16; ++d) w[d] = pool[((size_t)d * I + i) * O + o];
  const int n0 = blockIdx.y * 256;
  for (int n = n0; n < n0 + 256; ++n) {
    float acc = 0.f;
#pragma unroll
    for (int d = 0; d < 16; ++d) acc += emb[n * 16 + d] * w[d];
    Wout[((size_t)n * O + o) * I + i] = __float2bfloat16(acc);
  }
}

// bout[n][o] = sum_d E[n][d] * bpool[d][o]
__global__ void __launch_bounds__(256) k_bias(const float* __restrict__ emb,
                                              const float* __restrict__ bpool,
                                              float* __restrict__ bout, int O) {
  const int idx = blockIdx.x * 256 + threadIdx.x;
  const int n = idx / O, o = idx - n * O;
  float acc = 0.f;
#pragma unroll
  for (int d = 0; d < 16; ++d) acc += emb[n * 16 + d] * bpool[d * O + o];
  bout[idx] = acc;
}

// ---------------------------------------------------------------------------
// 4) big GEMM: C[r][c] = sum_k A[r][k] * Bt[c][k]   (4096x4096x4096, bf16)
//    128x128 block tile, 8 waves x (2x4) 16x16 WMMA tiles, K-chunks of 32.
//    Software-pipelined: fragments for chunk k+32 are in flight while the
//    8 WMMAs for chunk k execute (partial s_wait_loadcnt instead of 0).
// ---------------------------------------------------------------------------
__global__ void __launch_bounds__(256) k_gemm_nt(const bf16_t* __restrict__ A,
                                                 const bf16_t* __restrict__ Bt,
                                                 bf16_t* __restrict__ C) {
  const int K = 4096, Nc = 4096;
  const int r0 = blockIdx.y * 128;
  const int c0 = blockIdx.x * 128;
  const int wave = threadIdx.x >> 5;
  const int lane = threadIdx.x & 31;
  const int rt0 = (wave >> 1) * 2;   // 2 row tiles
  const int ct0 = (wave & 1) * 4;    // 4 col tiles

  const bf16_t* Ab[2];
  const bf16_t* Bb[4];
#pragma unroll
  for (int i = 0; i < 2; ++i) Ab[i] = A + (size_t)(r0 + (rt0 + i) * 16) * K;
#pragma unroll
  for (int j = 0; j < 4; ++j) Bb[j] = Bt + (size_t)(c0 + (ct0 + j) * 16) * K;

  const v8f zf = {0.f, 0.f, 0.f, 0.f, 0.f, 0.f, 0.f, 0.f};
  v8f acc[2][4];
#pragma unroll
  for (int i = 0; i < 2; ++i)
#pragma unroll
    for (int j = 0; j < 4; ++j) acc[i][j] = zf;

  bfrag a0[2], b0[4], a1[2], b1[4];
#pragma unroll
  for (int i = 0; i < 2; ++i) a0[i] = frag_load(Ab[i], K);
#pragma unroll
  for (int j = 0; j < 4; ++j) b0[j] = frag_load(Bb[j], K);

  // main loop: two chunks (kc via a0/b0, kc+32 via a1/b1) per iteration
  for (int kc = 0; kc < K - 64; kc += 64) {
#pragma unroll
    for (int i = 0; i < 2; ++i) a1[i] = frag_load(Ab[i] + kc + 32, K);
#pragma unroll
    for (int j = 0; j < 4; ++j) b1[j] = frag_load(Bb[j] + kc + 32, K);
#pragma unroll
    for (int i = 0; i < 2; ++i) frag_prefetch(Ab[i] + kc + 128, K);
#pragma unroll
    for (int j = 0; j < 4; ++j) frag_prefetch(Bb[j] + kc + 128, K);
#pragma unroll
    for (int i = 0; i < 2; ++i)
#pragma unroll
      for (int j = 0; j < 4; ++j)
        acc[i][j] = wmma_bf16(a0[i], b0[j], acc[i][j]);

#pragma unroll
    for (int i = 0; i < 2; ++i) a0[i] = frag_load(Ab[i] + kc + 64, K);
#pragma unroll
    for (int j = 0; j < 4; ++j) b0[j] = frag_load(Bb[j] + kc + 64, K);
#pragma unroll
    for (int i = 0; i < 2; ++i)
#pragma unroll
      for (int j = 0; j < 4; ++j)
        acc[i][j] = wmma_bf16(a1[i], b1[j], acc[i][j]);
  }

  // tail: chunks K-64 (already in a0/b0) and K-32
#pragma unroll
  for (int i = 0; i < 2; ++i) a1[i] = frag_load(Ab[i] + K - 32, K);
#pragma unroll
  for (int j = 0; j < 4; ++j) b1[j] = frag_load(Bb[j] + K - 32, K);
#pragma unroll
  for (int i = 0; i < 2; ++i)
#pragma unroll
    for (int j = 0; j < 4; ++j)
      acc[i][j] = wmma_bf16(a0[i], b0[j], acc[i][j]);
#pragma unroll
  for (int i = 0; i < 2; ++i)
#pragma unroll
    for (int j = 0; j < 4; ++j)
      acc[i][j] = wmma_bf16(a1[i], b1[j], acc[i][j]);

  const int col = lane & 15, half = lane >> 4;   // D: m = v + half*8, n = col
#pragma unroll
  for (int i = 0; i < 2; ++i)
#pragma unroll
    for (int j = 0; j < 4; ++j)
#pragma unroll
      for (int v = 0; v < 8; ++v) {
        const int rr = r0 + (rt0 + i) * 16 + v + half * 8;
        const int cc = c0 + (ct0 + j) * 16 + col;
        C[(size_t)rr * Nc + cc] = __float2bfloat16(acc[i][j][v]);
      }
}

// ---------------------------------------------------------------------------
// 5) gate: per node n, (64b x 128i) @ (128i x 128o) -> sigmoid -> z,r
//    A rows come straight from L2-resident rows of Sx / Sstate.
// ---------------------------------------------------------------------------
__global__ void __launch_bounds__(256) k_gate(const bf16_t* __restrict__ Sx,
                                              const bf16_t* __restrict__ Sst,
                                              const bf16_t* __restrict__ Wg,
                                              const float* __restrict__ biasg,
                                              const float* __restrict__ state,
                                              float* __restrict__ zs,
                                              float* __restrict__ rbuf) {
  const int n = blockIdx.x;
  const int wave = threadIdx.x >> 5;
  const int lane = threadIdx.x & 31;
  const int bt  = wave >> 1;         // batch tile 0..3
  const int ot0 = (wave & 1) * 4;    // 4 output tiles of 8

  const v8f zf = {0.f, 0.f, 0.f, 0.f, 0.f, 0.f, 0.f, 0.f};
  v8f acc[4];
#pragma unroll
  for (int j = 0; j < 4; ++j) acc[j] = zf;

  const bf16_t* Wn = Wg + (size_t)n * 128 * 128;
#pragma unroll
  for (int kk = 0; kk < 4; ++kk) {
    const int kc = kk * 32;
    const bf16_t* asrc = (kc < 64) ? Sx : Sst;     // concat([S@x, S@state])
    bfrag a = frag_load(asrc + (size_t)n * 4096 + (bt * 16) * 64 + (kc & 63), 64);
#pragma unroll
    for (int j = 0; j < 4; ++j) {
      bfrag b = frag_load(Wn + (size_t)((ot0 + j) * 16) * 128 + kc, 128);
      acc[j] = wmma_bf16(a, b, acc[j]);
    }
  }

  const int col = lane & 15, half = lane >> 4;
#pragma unroll
  for (int j = 0; j < 4; ++j) {
    const int o = (ot0 + j) * 16 + col;
    const float bias = biasg[n * 128 + o];
#pragma unroll
    for (int v = 0; v < 8; ++v) {
      const int b_ = bt * 16 + v + half * 8;
      const float s = 1.f / (1.f + __expf(-(acc[j][v] + bias)));
      const size_t idx = ((size_t)b_ * 4096 + n) * 64;
      if (o < 64) zs[idx + o] = s * state[idx + o];     // z * state
      else        rbuf[idx + (o - 64)] = s;             // r
    }
  }
}

// ---------------------------------------------------------------------------
// 6) update: per node n, (64b x 128i) @ (128i x 64o) -> tanh -> GRU combine
// ---------------------------------------------------------------------------
__global__ void __launch_bounds__(256) k_update(const bf16_t* __restrict__ Sx,
                                                const bf16_t* __restrict__ Szs,
                                                const bf16_t* __restrict__ Wu,
                                                const float* __restrict__ biasu,
                                                const float* __restrict__ state,
                                                const float* __restrict__ rbuf,
                                                float* __restrict__ out) {
  const int n = blockIdx.x;
  const int wave = threadIdx.x >> 5;
  const int lane = threadIdx.x & 31;
  const int bt  = wave >> 1;
  const int ot0 = (wave & 1) * 2;    // 2 output tiles of 4

  const v8f zf = {0.f, 0.f, 0.f, 0.f, 0.f, 0.f, 0.f, 0.f};
  v8f acc[2];
#pragma unroll
  for (int j = 0; j < 2; ++j) acc[j] = zf;

  const bf16_t* Wn = Wu + (size_t)n * 64 * 128;
#pragma unroll
  for (int kk = 0; kk < 4; ++kk) {
    const int kc = kk * 32;
    const bf16_t* asrc = (kc < 64) ? Sx : Szs;     // concat([S@x, S@(z*state)])
    bfrag a = frag_load(asrc + (size_t)n * 4096 + (bt * 16) * 64 + (kc & 63), 64);
#pragma unroll
    for (int j = 0; j < 2; ++j) {
      bfrag b = frag_load(Wn + (size_t)((ot0 + j) * 16) * 128 + kc, 128);
      acc[j] = wmma_bf16(a, b, acc[j]);
    }
  }

  const int col = lane & 15, half = lane >> 4;
#pragma unroll
  for (int j = 0; j < 2; ++j) {
    const int o = (ot0 + j) * 16 + col;
    const float bias = biasu[n * 64 + o];
#pragma unroll
    for (int v = 0; v < 8; ++v) {
      const int b_ = bt * 16 + v + half * 8;
      const float hc = tanhf(acc[j][v] + bias);
      const size_t idx = ((size_t)b_ * 4096 + n) * 64 + o;
      const float r = rbuf[idx];
      out[idx] = r * state[idx] + (1.f - r) * hc;
    }
  }
}

// ---------------------------------------------------------------------------
extern "C" void kernel_launch(void* const* d_in, const int* in_sizes, int n_in,
                              void* d_out, int out_size, void* d_ws, size_t ws_size,
                              hipStream_t stream) {
  (void)in_sizes; (void)n_in; (void)out_size; (void)ws_size;
  const float* x     = (const float*)d_in[0];   // (64,4096,64)
  const float* state = (const float*)d_in[1];   // (64,4096,64)
  const float* emb   = (const float*)d_in[2];   // (4096,16)
  const float* gwp   = (const float*)d_in[3];   // (16,128,128)
  const float* gbp   = (const float*)d_in[4];   // (16,128)
  const float* uwp   = (const float*)d_in[5];   // (16,128,64)
  const float* ubp   = (const float*)d_in[6];   // (16,64)
  float* out = (float*)d_out;                   // (64,4096,64)

  char* w = (char*)d_ws;
  const size_t MB = 1ull << 20;
  bf16_t* S    = (bf16_t*)(w + 0 * MB);     // 32MB  supports bf16
  bf16_t* Sx   = (bf16_t*)(w + 32 * MB);    // 32MB  S @ xT
  bf16_t* Sst  = (bf16_t*)(w + 64 * MB);    // 32MB  S @ stateT
  bf16_t* xT   = (bf16_t*)(w + 96 * MB);    // 32MB  x transposed/bf16
  bf16_t* stT  = (bf16_t*)(w + 128 * MB);   // 32MB  state transposed/bf16
  float*  zs   = (float*)(w + 96 * MB);     // 64MB  z*state  (reuses xT+stT)
  float*  rbuf = (float*)(w + 160 * MB);    // 64MB  r gate
  float*  bg   = (float*)(w + 224 * MB);    // 2MB   gate bias per node
  float*  bu   = (float*)(w + 226 * MB);    // 1MB   update bias per node
  bf16_t* Wg   = (bf16_t*)(w + 228 * MB);   // 128MB gate weights per node
  bf16_t* Wu   = (bf16_t*)(w + 356 * MB);   // 64MB  update weights per node
  bf16_t* zsT  = (bf16_t*)(w + 228 * MB);   // 32MB  (reuses Wg after gate)
  bf16_t* Szs  = (bf16_t*)(w + 260 * MB);   // 32MB  (reuses Wg after gate)

  k_supports<<<4096, 256, 0, stream>>>(emb, S);
  k_pack<<<dim3(64, 64), 256, 0, stream>>>(x, xT);
  k_pack<<<dim3(64, 64), 256, 0, stream>>>(state, stT);
  k_wexpand<<<dim3(64, 16), 256, 0, stream>>>(emb, gwp, Wg, 128, 128);
  k_wexpand<<<dim3(32, 16), 256, 0, stream>>>(emb, uwp, Wu, 128, 64);
  k_bias<<<2048, 256, 0, stream>>>(emb, gbp, bg, 128);
  k_bias<<<1024, 256, 0, stream>>>(emb, ubp, bu, 64);
  k_gemm_nt<<<dim3(32, 32), 256, 0, stream>>>(S, xT, Sx);     // S @ x
  k_gemm_nt<<<dim3(32, 32), 256, 0, stream>>>(S, stT, Sst);   // S @ state
  k_gate<<<4096, 256, 0, stream>>>(Sx, Sst, Wg, bg, state, zs, rbuf);
  k_pack<<<dim3(64, 64), 256, 0, stream>>>(zs, zsT);
  k_gemm_nt<<<dim3(32, 32), 256, 0, stream>>>(S, zsT, Szs);   // S @ (z*state)
  k_update<<<4096, 256, 0, stream>>>(Sx, Szs, Wu, bu, state, rbuf, out);
}
